// ShiftedWindowAttention_24094766530735
// MI455X (gfx1250) — matrix operand order repaired and checked by reading, not compiled, not run on hardware
//
#include <hip/hip_runtime.h>

// Swin shifted-window attention, fully fused per 7x7 window.
// B=32, H=W=56, C=256, heads=8, hd=32, N=49, nW=64 -> 2048 windows.

#define NPOS 49

typedef __attribute__((ext_vector_type(16))) __bf16 v16bf;
typedef __attribute__((ext_vector_type(8)))  float  v8f;

union Frag { unsigned u[8]; v16bf v; };

__device__ __forceinline__ unsigned short f2bf(float f) {
  unsigned u = __float_as_uint(f);
  u += 0x7FFFu + ((u >> 16) & 1u);     // round-to-nearest-even
  return (unsigned short)(u >> 16);
}
__device__ __forceinline__ unsigned pack2(float lo, float hi) {
  return (unsigned)f2bf(lo) | ((unsigned)f2bf(hi) << 16);
}
__device__ __forceinline__ v8f wmma_bf16(const Frag& a, const Frag& b, v8f c) {
  return __builtin_amdgcn_wmma_f32_16x16x32_bf16(false, a.v, false, b.v,
                                                 (short)0, c, false, false);
}
// Swin shift-mask region id for position p (0..63) in window (wh,ww)
__device__ __forceinline__ int regionCnt(int p, int wh, int ww) {
  int i = p / 7, j = p - i * 7;
  int gh = wh * 7 + i, gw = ww * 7 + j;
  int rh = (gh < 49) ? 0 : ((gh < 53) ? 1 : 2);
  int rw = (gw < 49) ? 0 : ((gw < 53) ? 1 : 2);
  return rh * 3 + rw;
}

// ---- prep: bf16-convert both weights; expand rel bias to dense [8,49,49] ----
__global__ void prep_kernel(const float* __restrict__ qkvw,
                            const float* __restrict__ projw,
                            const float* __restrict__ table,
                            const int*   __restrict__ relidx,
                            unsigned short* __restrict__ wq,
                            unsigned short* __restrict__ wp,
                            float* __restrict__ biasFull) {
  int idx = blockIdx.x * 256 + threadIdx.x;
  if (idx < 196608) {                       // 768*256
    wq[idx] = f2bf(qkvw[idx]);
  } else if (idx < 262144) {                // + 256*256
    wp[idx - 196608] = f2bf(projw[idx - 196608]);
  } else if (idx < 281352) {                // + 8*49*49
    int t = idx - 262144;
    int h = t / 2401, rem = t - h * 2401;
    biasFull[t] = table[relidx[rem] * 8 + h];
  }
}

// ---- fused per-window kernel: 256 threads = 8 wave32, one window per block ----
__global__ __launch_bounds__(256) void swin_kernel(
    const float* __restrict__ x, const float* __restrict__ qkvb,
    const float* __restrict__ projb, const unsigned* __restrict__ wq32,
    const unsigned* __restrict__ wp32, const float* __restrict__ biasFull,
    float* __restrict__ out) {
  extern __shared__ char smem[];
  unsigned short* AsS   = (unsigned short*)smem;            // 64x256 bf16 x-window, later out_win
  unsigned*       As32  = (unsigned*)smem;
  unsigned short* qkvS  = (unsigned short*)(smem + 32768);  // 64x768 bf16 qkv
  unsigned*       qkv32 = (unsigned*)(smem + 32768);
  unsigned short* atS   = (unsigned short*)(smem + 131072); // 8 waves x 64x64 bf16 attn
  unsigned*       at32  = (unsigned*)(smem + 131072);

  const int tid  = threadIdx.x;
  const int wave = tid >> 5;
  const int lane = tid & 31;
  const int half = lane >> 4;   // A/B K-half select
  const int l15  = lane & 15;

  const int widx = blockIdx.x;
  const int b   = widx >> 6;
  const int wi  = widx & 63;
  const int wh  = wi >> 3;
  const int wwc = wi & 7;

  // ---------- Phase 1: gather shifted window into LDS bf16 [64x256], pad rows 49..63 = 0
  for (int i = tid; i < 64 * 64; i += 256) {
    int row = i >> 6;
    int q4  = i & 63;
    float4 v = make_float4(0.f, 0.f, 0.f, 0.f);
    if (row < NPOS) {
      int ii = row / 7, jj = row - ii * 7;
      int hs = (wh * 7 + ii + 3) % 56;            // roll(x, -3)
      int ws2 = (wwc * 7 + jj + 3) % 56;
      const float4* xv = (const float4*)(x + (((size_t)b * 56 + hs) * 56 + ws2) * 256);
      v = xv[q4];
    }
    As32[row * 128 + q4 * 2]     = pack2(v.x, v.y);
    As32[row * 128 + q4 * 2 + 1] = pack2(v.z, v.w);
  }
  __syncthreads();

  // ---------- Phase 2: qkv = Xwin @ Wqkv^T + b (q pre-scaled) -> LDS bf16 [64x768]
  for (int t = wave; t < 192; t += 8) {           // 4 M-tiles x 48 N-tiles
    int mt = t & 3;
    int nt = t >> 2;
    int j  = nt * 16 + l15;                        // output column 0..767
    v8f acc = {0.f,0.f,0.f,0.f,0.f,0.f,0.f,0.f};
#pragma unroll
    for (int kt = 0; kt < 8; ++kt) {
      Frag a, bb;
      int m = mt * 16 + l15;
#pragma unroll
      for (int i = 0; i < 8; ++i)
        a.u[i] = As32[m * 128 + kt * 16 + ((i >= 4) ? 8 : 0) + half * 4 + (i & 3)];
#pragma unroll
      for (int vv = 0; vv < 8; ++vv)
        bb.u[vv] = wq32[j * 128 + kt * 16 + ((vv >= 4) ? 8 : 0) + half * 4 + (vv & 3)];
      acc = wmma_bf16(a, bb, acc);
    }
    float bia = qkvb[j];
    float scl = (j < 256) ? 0.17677669529663687f : 1.0f;   // hd^-0.5 on q
#pragma unroll
    for (int r = 0; r < 8; ++r) {
      int m = mt * 16 + half * 8 + r;
      qkvS[m * 768 + j] = f2bf((acc[r] + bia) * scl);
    }
  }
  __syncthreads();

  // ---------- Phase 3: attention, one head per wave
  {
    const int head = wave;
    const float* bp = biasFull + head * 2401;
    float sc[4][4][8];
    int ccnt[4];
#pragma unroll
    for (int nt = 0; nt < 4; ++nt) ccnt[nt] = regionCnt(nt * 16 + l15, wh, wwc);

#pragma unroll
    for (int mt = 0; mt < 4; ++mt) {
      Frag a;                                      // q fragment, K = hd = 32
      int m0 = mt * 16 + l15;
#pragma unroll
      for (int i = 0; i < 8; ++i)
        a.u[i] = qkv32[m0 * 384 + head * 16 + ((i >= 4) ? 8 : 0) + half * 4 + (i & 3)];
#pragma unroll
      for (int nt = 0; nt < 4; ++nt) {
        Frag bb;                                   // k^T fragment: B(d,n) = K[n][d]
        int n = nt * 16 + l15;
#pragma unroll
        for (int vv = 0; vv < 8; ++vv)
          bb.u[vv] = qkv32[n * 384 + 128 + head * 16 + ((vv >= 4) ? 8 : 0) + half * 4 + (vv & 3)];
        v8f c = {0.f,0.f,0.f,0.f,0.f,0.f,0.f,0.f};
        c = wmma_bf16(a, bb, c);
#pragma unroll
        for (int r = 0; r < 8; ++r) {
          int m = mt * 16 + half * 8 + r;
          float val;
          if (n < NPOS) {
            val = c[r];
            if (m < NPOS) val += bp[m * 49 + n];
            val += (regionCnt(m, wh, wwc) == ccnt[nt]) ? 0.f : -100.f;
          } else {
            val = -1e30f;                          // pad key columns
          }
          sc[mt][nt][r] = val;
        }
      }
    }
    // row softmax: row lives in 4 regs (nt) x 16 lanes of this half
#pragma unroll
    for (int mt = 0; mt < 4; ++mt) {
#pragma unroll
      for (int r = 0; r < 8; ++r) {
        float mx = sc[mt][0][r];
#pragma unroll
        for (int nt = 1; nt < 4; ++nt) mx = fmaxf(mx, sc[mt][nt][r]);
#pragma unroll
        for (int off = 1; off < 16; off <<= 1) mx = fmaxf(mx, __shfl_xor(mx, off, 32));
        float sum = 0.f;
#pragma unroll
        for (int nt = 0; nt < 4; ++nt) {
          float e = __expf(sc[mt][nt][r] - mx);
          sc[mt][nt][r] = e; sum += e;
        }
#pragma unroll
        for (int off = 1; off < 16; off <<= 1) sum += __shfl_xor(sum, off, 32);
        float inv = __builtin_amdgcn_rcpf(sum);
#pragma unroll
        for (int nt = 0; nt < 4; ++nt) sc[mt][nt][r] *= inv;
      }
    }
    // stage attn bf16 into per-wave LDS scratch [64x64]
#pragma unroll
    for (int mt = 0; mt < 4; ++mt)
#pragma unroll
      for (int nt = 0; nt < 4; ++nt)
#pragma unroll
        for (int r = 0; r < 8; ++r) {
          int m = mt * 16 + half * 8 + r;
          int n = nt * 16 + l15;
          atS[wave * 4096 + m * 64 + n] = f2bf(sc[mt][nt][r]);
        }
  }
  __syncthreads();

  // ---------- Phase 3b: out_head = attn @ V -> out_win (reuse As region)
  {
    const int head = wave;
#pragma unroll
    for (int mt = 0; mt < 4; ++mt) {
#pragma unroll
      for (int nt2 = 0; nt2 < 2; ++nt2) {
        v8f c = {0.f,0.f,0.f,0.f,0.f,0.f,0.f,0.f};
#pragma unroll
        for (int kt = 0; kt < 2; ++kt) {           // K = 64 (padded), 2 x K32 steps
          Frag a, bb;
          int m0 = mt * 16 + l15;
#pragma unroll
          for (int i = 0; i < 8; ++i)
            a.u[i] = at32[wave * 2048 + m0 * 32 + kt * 16 + ((i >= 4) ? 8 : 0) + half * 4 + (i & 3)];
#pragma unroll
          for (int vv = 0; vv < 8; ++vv) {         // B(k,n) = V[k][n]; K-pairs strided -> u16 gather
            int n  = nt2 * 16 + l15;
            int k0 = kt * 32 + ((vv >= 4) ? 16 : 0) + half * 8 + (vv & 3) * 2;
            unsigned lo = qkvS[k0 * 768 + 512 + head * 32 + n];
            unsigned hi = qkvS[(k0 + 1) * 768 + 512 + head * 32 + n];
            bb.u[vv] = lo | (hi << 16);
          }
          c = wmma_bf16(a, bb, c);
        }
#pragma unroll
        for (int r = 0; r < 8; ++r) {
          int m = mt * 16 + half * 8 + r;
          if (m < NPOS)
            AsS[m * 256 + head * 32 + nt2 * 16 + l15] = f2bf(c[r]);
        }
      }
    }
  }
  __syncthreads();

  // ---------- Phase 4: proj GEMM + bias + scatter with inverse roll
  for (int t = wave; t < 64; t += 8) {             // 4 M-tiles x 16 N-tiles
    int mt  = t & 3;
    int ntj = t >> 2;
    int j   = ntj * 16 + l15;                      // 0..255
    v8f acc = {0.f,0.f,0.f,0.f,0.f,0.f,0.f,0.f};
#pragma unroll
    for (int kt = 0; kt < 8; ++kt) {
      Frag a, bb;
      int m = mt * 16 + l15;
#pragma unroll
      for (int i = 0; i < 8; ++i)
        a.u[i] = As32[m * 128 + kt * 16 + ((i >= 4) ? 8 : 0) + half * 4 + (i & 3)];
#pragma unroll
      for (int vv = 0; vv < 8; ++vv)
        bb.u[vv] = wp32[j * 128 + kt * 16 + ((vv >= 4) ? 8 : 0) + half * 4 + (vv & 3)];
      acc = wmma_bf16(a, bb, acc);
    }
    float bia = projb[j];
#pragma unroll
    for (int r = 0; r < 8; ++r) {
      int m = mt * 16 + half * 8 + r;
      if (m < NPOS) {
        int ii = m / 7, jj = m - ii * 7;
        int hd2 = (wh * 7 + ii + 3) % 56;          // roll(out, +3)
        int wd2 = (wwc * 7 + jj + 3) % 56;
        out[(((size_t)b * 56 + hd2) * 56 + wd2) * 256 + j] = acc[r] + bia;
      }
    }
  }
}

extern "C" void kernel_launch(void* const* d_in, const int* in_sizes, int n_in,
                              void* d_out, int out_size, void* d_ws, size_t ws_size,
                              hipStream_t stream) {
  const float* x     = (const float*)d_in[0];
  const float* qkvw  = (const float*)d_in[1];
  const float* qkvb  = (const float*)d_in[2];
  const float* projw = (const float*)d_in[3];
  const float* projb = (const float*)d_in[4];
  const float* table = (const float*)d_in[5];
  const int*   ridx  = (const int*)d_in[6];
  float* out = (float*)d_out;

  // workspace layout (bytes): [0, 393216) qkv_w bf16; [393216, 524288) proj_w bf16;
  // [524288, 601120) dense rel-pos bias f32 [8,49,49]
  unsigned short* wq = (unsigned short*)d_ws;
  unsigned short* wp = (unsigned short*)((char*)d_ws + 393216);
  float* biasFull    = (float*)((char*)d_ws + 524288);

  prep_kernel<<<1100, 256, 0, stream>>>(qkvw, projw, table, ridx, wq, wp, biasFull);

  (void)hipFuncSetAttribute((const void*)swin_kernel,
                            hipFuncAttributeMaxDynamicSharedMemorySize, 196608);
  swin_kernel<<<2048, 256, 196608, stream>>>(x, qkvb, projb,
                                             (const unsigned*)wq, (const unsigned*)wp,
                                             biasFull, out);
}
